// SpeakerEncoderLSTM_29463475651099
// MI455X (gfx1250) — matrix operand order repaired and checked by reading, not compiled, not run on hardware
//
#include <hip/hip_runtime.h>
#include <hip/hip_bf16.h>
#include <math.h>

// ---------------- problem constants (from reference) ----------------
#define TT 40
#define BB 64
#define VN 36
#define A_DIM 2048
#define V_DIM 2048
#define HH 1024
#define DOT 256
#define G4H 4096   // 4*H

typedef __attribute__((ext_vector_type(16))) __bf16 v16bf;
typedef __attribute__((ext_vector_type(8)))  float  v8f;
typedef int v4i_ __attribute__((vector_size(4 * sizeof(int))));

// ---------------- gfx1250 async global->LDS support ----------------
#if defined(__has_builtin)
# if __has_builtin(__builtin_amdgcn_global_load_async_to_lds_b128)
#  define HAVE_ASYNC_LDS 1
# endif
#endif
#ifndef HAVE_ASYNC_LDS
# define HAVE_ASYNC_LDS 0
# warning "gfx1250 async global->LDS builtin NOT available; falling back to synchronous LDS staging"
#endif

#if HAVE_ASYNC_LDS
# if __has_builtin(__builtin_amdgcn_s_wait_asynccnt)
#  define WAIT_ASYNC(n) __builtin_amdgcn_s_wait_asynccnt(n)
# else
#  define WAIT_ASYNC(n) asm volatile("s_wait_asynccnt %0" ::"n"(n) : "memory")
# endif
#else
# define WAIT_ASYNC(n) ((void)0)
#endif

__device__ __forceinline__ void stage16(const float* g, float* l)
{
#if HAVE_ASYNC_LDS
    __builtin_amdgcn_global_load_async_to_lds_b128((v4i_*)g, (v4i_*)l, 0, 0);
#else
    *(float4*)l = *(const float4*)g;
#endif
}

// --------------------------------------------------------------------
// Cooperative bf16-WMMA GEMM:  C = A(MxK, rm) * op(B) + Cin + bias
//   BT=true : B is NxK row-major (A*B^T), contiguous staging
//   BT=false: B is KxN row-major (A*B),   staged as [k][n]
// Block tile 64x128 (8 waves, each 32x32 = 2x2 WMMA tiles).
// K consumed in 32-chunks, double-buffered in LDS via async copies.
// Requires M%64==0, N%128==0, K%32==0 (true for all calls below).
//
// WMMA wave32 operand layouts as loaded here (A and B share one contiguous k order):
//   A 16x32 bf16 : lane<16 -> M=lane,    K=0..15
//                  lane>=16 -> M=lane-16, K=16..31
//   B 32x16 bf16 : lane<16 -> N=lane,    K=0..15 ; lane>=16 -> K=16..31
//   C/D 16x16 f32: VGPR r -> (M = r + 8*(lane>=16), N = lane&15)
// --------------------------------------------------------------------
template <bool BT>
__global__ __launch_bounds__(256)
void gemm_wmma_bf16(const float* __restrict__ A, int lda,
                    const float* __restrict__ B, int ldb,
                    const float* __restrict__ Cin,
                    const float* __restrict__ bias,
                    float* __restrict__ Cout, int ldc,
                    int M, int N, int K)
{
    constexpr int BS_ELEMS = BT ? 128 * 36 : 32 * 132;
    __shared__ float As[2][64 * 36];
    __shared__ float Bs[2][BS_ELEMS];

    const int tid  = threadIdx.x;
    const int wid  = tid >> 5;
    const int lane = tid & 31;
    const int half = lane >> 4;      // 0/1
    const int l    = lane & 15;
    const int wm   = wid >> 2;       // 0..1  (rows within block tile)
    const int wn   = wid & 3;        // 0..3  (cols within block tile)
    const int gn   = N >> 7;
    const int bm   = blockIdx.x / gn;
    const int bn   = blockIdx.x % gn;

    auto stage = [&](int k0, int buf) {
        // A chunk: 64 rows x 32 floats = 512 x16B segments, 2 per thread
#pragma unroll
        for (int s = 0; s < 2; ++s) {
            const int seg = tid + s * 256;
            const int ar = seg >> 3, ak = (seg & 7) * 4;
            stage16(A + (size_t)(bm * 64 + ar) * lda + k0 + ak,
                    &As[buf][ar * 36 + ak]);
        }
        if constexpr (BT) {
            // B chunk: 128 N-rows x 32 floats = 1024 segments, 4 per thread
#pragma unroll
            for (int s = 0; s < 4; ++s) {
                const int seg = tid + s * 256;
                const int br = seg >> 3, bk = (seg & 7) * 4;
                stage16(B + (size_t)(bn * 128 + br) * ldb + k0 + bk,
                        &Bs[buf][br * 36 + bk]);
            }
        } else {
            // B chunk: 32 k-rows x 128 floats, staged [k][n]
#pragma unroll
            for (int s = 0; s < 4; ++s) {
                const int seg = tid + s * 256;
                const int br = seg >> 5, bc = (seg & 31) * 4;
                stage16(B + (size_t)(k0 + br) * ldb + bn * 128 + bc,
                        &Bs[buf][br * 132 + bc]);
            }
        }
    };

    v8f acc[2][2] = {};
    const int nch = K >> 5;

    stage(0, 0);
    for (int i = 0; i < nch; ++i) {
        if (i + 1 < nch) { stage((i + 1) << 5, (i + 1) & 1); WAIT_ASYNC(6); }
        else             { WAIT_ASYNC(0); }
        __syncthreads();

        const int buf = i & 1;
        const float* Ac = As[buf];
        const float* Bc = Bs[buf];

        v16bf afr[2];
#pragma unroll
        for (int ti = 0; ti < 2; ++ti) {
            const float* ap = Ac + (wm * 32 + ti * 16 + l) * 36 + half * 16;
            const float4 f0 = *(const float4*)(ap);
            const float4 f1 = *(const float4*)(ap + 4);
            const float4 f2 = *(const float4*)(ap + 8);
            const float4 f3 = *(const float4*)(ap + 12);
            const float av[16] = {f0.x, f0.y, f0.z, f0.w, f1.x, f1.y, f1.z, f1.w,
                                  f2.x, f2.y, f2.z, f2.w, f3.x, f3.y, f3.z, f3.w};
#pragma unroll
            for (int j = 0; j < 16; ++j) afr[ti][j] = (__bf16)av[j];
        }

        v16bf bfr[2];
#pragma unroll
        for (int tj = 0; tj < 2; ++tj) {
            float bv[16];
            if constexpr (BT) {
                const float* bp = Bc + (wn * 32 + tj * 16 + l) * 36 + half * 16;
#pragma unroll
                for (int j = 0; j < 16; ++j) bv[j] = bp[j];
            } else {
                const int col = wn * 32 + tj * 16 + l;
#pragma unroll
                for (int j = 0; j < 16; ++j) bv[j] = Bc[(half * 16 + j) * 132 + col];
            }
#pragma unroll
            for (int j = 0; j < 16; ++j) bfr[tj][j] = (__bf16)bv[j];
        }

#pragma unroll
        for (int ti = 0; ti < 2; ++ti)
#pragma unroll
            for (int tj = 0; tj < 2; ++tj)
                acc[ti][tj] = __builtin_amdgcn_wmma_f32_16x16x32_bf16(
                    false, afr[ti], false, bfr[tj], (short)0, acc[ti][tj],
                    false, false);
        __syncthreads();
    }

    // ---- epilogue ----
#pragma unroll
    for (int ti = 0; ti < 2; ++ti)
#pragma unroll
        for (int tj = 0; tj < 2; ++tj)
#pragma unroll
            for (int r = 0; r < 8; ++r) {
                const int row = bm * 64 + wm * 32 + ti * 16 + half * 8 + r;
                const int col = bn * 128 + wn * 32 + tj * 16 + l;
                const size_t idx = (size_t)row * ldc + col;
                float v = acc[ti][tj][r];
                if (Cin)  v += Cin[idx];
                if (bias) v += bias[col];
                Cout[idx] = v;
            }
}

// --------------------------------------------------------------------
// Attention: logits[b,n] = <world[t,b,n,:], u[b,:]> ; softmax over n ;
//            feature[b,:] = sum_n attn[n] * world[t,b,n,:]
// --------------------------------------------------------------------
__global__ __launch_bounds__(256)
void attn_kernel(const float* __restrict__ world_t,  // [B, VN, V_DIM]
                 const float* __restrict__ u,        // [B, V_DIM]
                 float* __restrict__ feature)        // [B, V_DIM]
{
    const int b   = blockIdx.x;
    const int tid = threadIdx.x;
    __shared__ float part[8][VN];
    __shared__ float attn_s[VN];

    const float* wb = world_t + (size_t)b * VN * V_DIM;
    const float* ub = u + (size_t)b * V_DIM + tid * 8;

    float ur[8];
#pragma unroll
    for (int j = 0; j < 8; ++j) ur[j] = ub[j];

    const int w = tid >> 5, lane = tid & 31;
    for (int n = 0; n < VN; ++n) {
        const float* wn = wb + (size_t)n * V_DIM + tid * 8;
        float p = 0.f;
#pragma unroll
        for (int j = 0; j < 8; ++j) p += wn[j] * ur[j];
#pragma unroll
        for (int off = 16; off > 0; off >>= 1) p += __shfl_xor(p, off, 32);
        if (lane == 0) part[w][n] = p;
    }
    __syncthreads();

    if (tid == 0) {
        float mx = -INFINITY;
        for (int n = 0; n < VN; ++n) {
            float s = 0.f;
            for (int k = 0; k < 8; ++k) s += part[k][n];
            attn_s[n] = s;
            mx = fmaxf(mx, s);
        }
        float sum = 0.f;
        for (int n = 0; n < VN; ++n) {
            float e = __expf(attn_s[n] - mx);
            attn_s[n] = e;
            sum += e;
        }
        const float inv = 1.f / sum;
        for (int n = 0; n < VN; ++n) attn_s[n] *= inv;
    }
    __syncthreads();

    float accf[8] = {0.f, 0.f, 0.f, 0.f, 0.f, 0.f, 0.f, 0.f};
    for (int n = 0; n < VN; ++n) {
        const float an = attn_s[n];
        const float* wn = wb + (size_t)n * V_DIM + tid * 8;
#pragma unroll
        for (int j = 0; j < 8; ++j) accf[j] += an * wn[j];
    }
    float* fo = feature + (size_t)b * V_DIM + tid * 8;
#pragma unroll
    for (int j = 0; j < 8; ++j) fo[j] = accf[j];
}

// --------------------------------------------------------------------
__device__ __forceinline__ float sigf(float x) { return 1.f / (1.f + __expf(-x)); }

__global__ __launch_bounds__(256)
void lstm_kernel(const float* __restrict__ gates,  // [B, 4H] order i,f,g,o
                 float* __restrict__ h, float* __restrict__ c,
                 float* __restrict__ ctx_out,      // [B, T, H]
                 int t)
{
    const int idx = blockIdx.x * 256 + threadIdx.x;   // 0..B*H-1
    const int b  = idx >> 10;
    const int hi = idx & (HH - 1);
    const float* g = gates + (size_t)b * G4H;
    const float gi = g[hi];
    const float gf = g[HH + hi];
    const float gg = g[2 * HH + hi];
    const float go = g[3 * HH + hi];
    const float c1 = sigf(gf) * c[idx] + sigf(gi) * tanhf(gg);
    const float h1 = sigf(go) * tanhf(c1);
    c[idx] = c1;
    h[idx] = h1;
    ctx_out[((size_t)b * TT + t) * HH + hi] = h1;
}

__global__ __launch_bounds__(256)
void zero_kernel(float* __restrict__ p, int n)
{
    const int i = blockIdx.x * 256 + threadIdx.x;
    if (i < n) p[i] = 0.f;
}

__global__ __launch_bounds__(256)
void finish_kernel(const float* __restrict__ dtmp, const float* __restrict__ c,
                   float* __restrict__ out_dec, float* __restrict__ out_c)
{
    const int idx = blockIdx.x * 256 + threadIdx.x;   // 0..B*H-1
    out_dec[idx] = tanhf(dtmp[idx]);
    out_c[idx]   = c[idx];
}

// --------------------------------------------------------------------
extern "C" void kernel_launch(void* const* d_in, const int* in_sizes, int n_in,
                              void* d_out, int out_size, void* d_ws, size_t ws_size,
                              hipStream_t stream)
{
    const float* act   = (const float*)d_in[0];   // [T,B,A_DIM]
    const float* world = (const float*)d_in[1];   // [T,B,VN,V_DIM]
    const float* W_h   = (const float*)d_in[2];   // [DOT,H]
    const float* b_h   = (const float*)d_in[3];   // [DOT]
    const float* W_v   = (const float*)d_in[4];   // [DOT,V_DIM]
    // d_in[5] = b_v : adds a per-row constant to logits -> cancels in softmax
    const float* W_ih  = (const float*)d_in[6];   // [4H, A_DIM+V_DIM]
    const float* b_ih  = (const float*)d_in[7];   // [4H]
    const float* W_hh  = (const float*)d_in[8];   // [4H, H]
    const float* b_hh  = (const float*)d_in[9];   // [4H]
    const float* W_e2d = (const float*)d_in[10];  // [H,H]
    const float* b_e2d = (const float*)d_in[11];  // [H]

    float* out = (float*)d_out;
    float* out_ctx = out;                          // [B,T,H]
    float* out_dec = out + (size_t)BB * TT * HH;   // [B,H]
    float* out_c   = out_dec + (size_t)BB * HH;    // [B,H]

    // ---- workspace layout (floats) ----
    float* ws = (float*)d_ws;
    float* gates_a = ws;                                   // [T*B, 4H]
    float* gtmp    = gates_a + (size_t)TT * BB * G4H;      // [B, 4H]
    float* h       = gtmp + (size_t)BB * G4H;              // [B, H]
    float* c       = h + (size_t)BB * HH;                  // [B, H]
    float* target  = c + (size_t)BB * HH;                  // [B, DOT]
    float* u       = target + (size_t)BB * DOT;            // [B, V_DIM]
    float* feat    = u + (size_t)BB * V_DIM;               // [B, V_DIM]
    float* dtmp    = feat + (size_t)BB * V_DIM;            // [B, H]

    // h0 = c0 = 0
    zero_kernel<<<(2 * BB * HH + 255) / 256, 256, 0, stream>>>(h, 2 * BB * HH);

    // gates_a[t,b,:] = a_t @ W_ih[:, :A_DIM]^T + b_ih  (parallel over all t)
    gemm_wmma_bf16<true><<<(TT * BB / 64) * (G4H / 128), 256, 0, stream>>>(
        act, A_DIM, W_ih, A_DIM + V_DIM, nullptr, b_ih,
        gates_a, G4H, TT * BB, G4H, A_DIM);

    for (int t = 0; t < TT; ++t) {
        const float* world_t = world + (size_t)t * BB * VN * V_DIM;

        // target = h @ W_h^T + b_h          [B,DOT]
        gemm_wmma_bf16<true><<<(BB / 64) * (DOT / 128), 256, 0, stream>>>(
            h, HH, W_h, HH, nullptr, b_h, target, DOT, BB, DOT, HH);

        // u = target @ W_v                  [B,V_DIM]  (B is KxN)
        gemm_wmma_bf16<false><<<(BB / 64) * (V_DIM / 128), 256, 0, stream>>>(
            target, DOT, W_v, V_DIM, nullptr, nullptr, u, V_DIM, BB, V_DIM, DOT);

        // attention -> feature              [B,V_DIM]
        attn_kernel<<<BB, 256, 0, stream>>>(world_t, u, feat);

        // gtmp = gates_a[t] + feature @ W_ih[:, A_DIM:]^T
        gemm_wmma_bf16<true><<<(BB / 64) * (G4H / 128), 256, 0, stream>>>(
            feat, V_DIM, W_ih + A_DIM, A_DIM + V_DIM,
            gates_a + (size_t)t * BB * G4H, nullptr,
            gtmp, G4H, BB, G4H, V_DIM);

        // gtmp += h @ W_hh^T + b_hh
        gemm_wmma_bf16<true><<<(BB / 64) * (G4H / 128), 256, 0, stream>>>(
            h, HH, W_hh, HH, gtmp, b_hh, gtmp, G4H, BB, G4H, HH);

        // LSTM pointwise; writes h,c and ctx[:,t,:]
        lstm_kernel<<<(BB * HH) / 256, 256, 0, stream>>>(gtmp, h, c, out_ctx, t);
    }

    // decoder_init = tanh(h_T @ W_e2d^T + b_e2d)
    gemm_wmma_bf16<true><<<(BB / 64) * (HH / 128), 256, 0, stream>>>(
        h, HH, W_e2d, HH, nullptr, b_e2d, dtmp, HH, BB, HH, HH);

    finish_kernel<<<(BB * HH) / 256, 256, 0, stream>>>(dtmp, c, out_dec, out_c);
}
